// CustomLSTMModel_45088566674124
// MI455X (gfx1250) — matrix operand order repaired and checked by reading, not compiled, not run on hardware
//
#include <hip/hip_runtime.h>

// ---------------------------------------------------------------------------
// LSTM forward for MI455X (gfx1250, wave32, WMMA).
// Strategy: bf16 WMMA (16x16x32) with f32 accumulation. Per timestep the fused
// GEMM [X_t | H] (64x1280) x [Wx;Wh] (1280x4096) is tiled 16x16 per wave with
// all four gates computed per tile (A-fragment reuse). Gate nonlinearities,
// C/H update and the H.Wd output projection are fused into the epilogue.
// Weights are re-packed once into WMMA B-fragment order in bf16 (lives in L2
// across all 512 steps: ~10 MB << 192 MB).
// ---------------------------------------------------------------------------

typedef __attribute__((ext_vector_type(16))) __bf16 v16bf;
typedef __attribute__((ext_vector_type(8)))  float  v8f;

#define S_SEQ 512
#define D_IN  256
#define H_DIM 1024
#define LDK   1280            // D_IN + H_DIM (A panel K extent)
#define KCX   8               // D_IN / 32  K-chunks
#define KCH   32              // H_DIM / 32 K-chunks
#define GSX   (64 * KCX * 512)   // per-gate Wx fragment elements (262144)
#define GSH   (64 * KCH * 512)   // per-gate Wh fragment elements (1048576)

union Frag {
  uint4 u[2];
  v16bf v;
};

__device__ __forceinline__ unsigned short f32_to_bf16(float f) {
  unsigned int u = __float_as_uint(f);
  u += 0x7FFFu + ((u >> 16) & 1u);   // round-to-nearest-even
  return (unsigned short)(u >> 16);
}

__device__ __forceinline__ unsigned int pack2_bf16(float lo, float hi) {
  return (unsigned int)f32_to_bf16(lo) | ((unsigned int)f32_to_bf16(hi) << 16);
}

__device__ __forceinline__ float sigmoid_f(float x) {
  return 1.0f / (1.0f + __expf(-x));
}

// ---------------------------------------------------------------------------
// Pack a (Ktot x 1024) f32 weight matrix (x4 gates) into bf16 WMMA B-fragment
// order: frag[((g*64 + n_tile)*KC + kc)*32 + lane][j] where
//   lane <  16 : N = n_tile*16 + lane,      K = kc*32 + (j<8 ? j : 8 + j)
//   lane >= 16 : N = n_tile*16 + (lane-16), K = kc*32 + 8 + (j<8 ? j : 8 + j)
// ---------------------------------------------------------------------------
__global__ void pack_b_kernel(const float* __restrict__ w0, const float* __restrict__ w1,
                              const float* __restrict__ w2, const float* __restrict__ w3,
                              unsigned short* __restrict__ dst, int KC, long total) {
  long i = (long)blockIdx.x * blockDim.x + threadIdx.x;
  if (i >= total) return;
  long r = i;
  int j    = (int)(r & 15);  r >>= 4;
  int lane = (int)(r & 31);  r >>= 5;
  int kc   = (int)(r % KC);  r /= KC;
  int nt   = (int)(r & 63);  r >>= 6;
  int g    = (int)r;
  const float* W = (g == 0) ? w0 : (g == 1) ? w1 : (g == 2) ? w2 : w3;
  int col = nt * 16 + (lane & 15);
  int k   = kc * 32 + ((lane >> 4) * 8 + (j < 8 ? j : 8 + j));
  dst[i] = f32_to_bf16(W[(size_t)k * H_DIM + col]);
}

// Copy initial state into workspace ping-pong buffers; seed pred with bd.
// Re-runs on every call/replay so accumulation into pred stays deterministic.
__global__ void init_state_kernel(const float* __restrict__ H0, const float* __restrict__ C0,
                                  const float* __restrict__ bd,
                                  float* __restrict__ Hb, float* __restrict__ Cb,
                                  float* __restrict__ pred) {
  int i = blockIdx.x * blockDim.x + threadIdx.x;
  if (i < 64 * H_DIM) { Hb[i] = H0[i]; Cb[i] = C0[i]; }
  if (i < 64 * S_SEQ) pred[i] = bd[0];
}

// ---------------------------------------------------------------------------
// One recurrent step. Grid: (16, 4) blocks of 128 threads (4 wave32).
//   blockIdx.y -> 16-row batch tile, blockIdx.x*4 + wave -> 16-col H tile.
// ---------------------------------------------------------------------------
__global__ __launch_bounds__(128)
void lstm_step_kernel(const float* __restrict__ X,
                      const unsigned short* __restrict__ wxF,
                      const unsigned short* __restrict__ whF,
                      const float* __restrict__ bI, const float* __restrict__ bF,
                      const float* __restrict__ bO, const float* __restrict__ bC,
                      const float* __restrict__ Wd,
                      const float* __restrict__ Hin, float* __restrict__ Hout,
                      float* __restrict__ C, float* __restrict__ pred, int t) {
  __shared__ __align__(16) unsigned short sA[16 * LDK];   // 40 KB bf16 A panel

  const int tid  = threadIdx.x;
  const int rowB = blockIdx.y * 16;

  // Stage X_t tile (16 x 256) -> LDS bf16 (packed: one b64 store per float4)
  for (int i = tid; i < 16 * (D_IN / 4); i += 128) {
    int m = i >> 6, c4 = i & 63;
    const float4 f = ((const float4*)(X + ((size_t)(rowB + m) * S_SEQ + t) * D_IN))[c4];
    uint2 pr;
    pr.x = pack2_bf16(f.x, f.y);
    pr.y = pack2_bf16(f.z, f.w);
    *(uint2*)&sA[m * LDK + c4 * 4] = pr;   // 8B-aligned: row pitch 2560B, off c4*8
  }
  // Stage H tile (16 x 1024) -> LDS bf16
  for (int i = tid; i < 16 * (H_DIM / 4); i += 128) {
    int m = i >> 8, c4 = i & 255;
    const float4 f = ((const float4*)(Hin + (size_t)(rowB + m) * H_DIM))[c4];
    uint2 pr;
    pr.x = pack2_bf16(f.x, f.y);
    pr.y = pack2_bf16(f.z, f.w);
    *(uint2*)&sA[m * LDK + D_IN + c4 * 4] = pr;
  }
  __syncthreads();

  const int wave = tid >> 5;
  const int lane = tid & 31;
  const int n    = blockIdx.x * 4 + wave;   // column tile 0..63
  const int ml   = lane & 15;
  const int hiL  = lane >> 4;

  v8f aI = {}, aF = {}, aO = {}, aC = {};

  // K-chunks over the X part (Wx fragments)
  const unsigned short* px = wxF + ((size_t)n * KCX * 32 + lane) * 16;
  #pragma unroll 2
  for (int kc = 0; kc < KCX; ++kc) {
    Frag a, f0, f1, f2, f3;
    const unsigned short* ar = &sA[ml * LDK + kc * 32 + hiL * 8];
    a.u[0] = *(const uint4*)ar;
    a.u[1] = *(const uint4*)(ar + 16);
    f0.u[0] = *(const uint4*)(px);             f0.u[1] = *(const uint4*)(px + 8);
    f1.u[0] = *(const uint4*)(px + GSX);       f1.u[1] = *(const uint4*)(px + GSX + 8);
    f2.u[0] = *(const uint4*)(px + 2 * GSX);   f2.u[1] = *(const uint4*)(px + 2 * GSX + 8);
    f3.u[0] = *(const uint4*)(px + 3 * GSX);   f3.u[1] = *(const uint4*)(px + 3 * GSX + 8);
    aI = __builtin_amdgcn_wmma_f32_16x16x32_bf16(false, a.v, false, f0.v, (short)0, aI, false, false);
    aF = __builtin_amdgcn_wmma_f32_16x16x32_bf16(false, a.v, false, f1.v, (short)0, aF, false, false);
    aO = __builtin_amdgcn_wmma_f32_16x16x32_bf16(false, a.v, false, f2.v, (short)0, aO, false, false);
    aC = __builtin_amdgcn_wmma_f32_16x16x32_bf16(false, a.v, false, f3.v, (short)0, aC, false, false);
    px += 512;
  }
  // K-chunks over the H part (Wh fragments)
  const unsigned short* ph = whF + ((size_t)n * KCH * 32 + lane) * 16;
  #pragma unroll 2
  for (int kc = 0; kc < KCH; ++kc) {
    Frag a, f0, f1, f2, f3;
    const unsigned short* ar = &sA[ml * LDK + D_IN + kc * 32 + hiL * 8];
    a.u[0] = *(const uint4*)ar;
    a.u[1] = *(const uint4*)(ar + 16);
    f0.u[0] = *(const uint4*)(ph);             f0.u[1] = *(const uint4*)(ph + 8);
    f1.u[0] = *(const uint4*)(ph + GSH);       f1.u[1] = *(const uint4*)(ph + GSH + 8);
    f2.u[0] = *(const uint4*)(ph + 2 * GSH);   f2.u[1] = *(const uint4*)(ph + 2 * GSH + 8);
    f3.u[0] = *(const uint4*)(ph + 3 * GSH);   f3.u[1] = *(const uint4*)(ph + 3 * GSH + 8);
    aI = __builtin_amdgcn_wmma_f32_16x16x32_bf16(false, a.v, false, f0.v, (short)0, aI, false, false);
    aF = __builtin_amdgcn_wmma_f32_16x16x32_bf16(false, a.v, false, f1.v, (short)0, aF, false, false);
    aO = __builtin_amdgcn_wmma_f32_16x16x32_bf16(false, a.v, false, f2.v, (short)0, aO, false, false);
    aC = __builtin_amdgcn_wmma_f32_16x16x32_bf16(false, a.v, false, f3.v, (short)0, aC, false, false);
    ph += 512;
  }

  // Epilogue: C/D layout -> element (M = v + hiL*8, N = lane&15) in acc[v].
  const int col = n * 16 + ml;
  const float biV = bI[col], bfV = bF[col], boV = bO[col], bcV = bC[col];
  const float wdV = Wd[col];

  #pragma unroll
  for (int v = 0; v < 8; ++v) {
    int b = rowB + v + hiL * 8;
    float gi = sigmoid_f(aI[v] + biV);
    float gf = sigmoid_f(aF[v] + bfV);
    float go = sigmoid_f(aO[v] + boV);
    float gc = tanhf(aC[v] + bcV);
    size_t off = (size_t)b * H_DIM + col;
    float cNew = gf * C[off] + gi * gc;
    C[off] = cNew;
    float hNew = go * tanhf(cNew);
    Hout[off] = hNew;
    // Partial of pred[b,t] = sum_h hNew * Wd[h]: reduce over the 16 lanes
    // that share this M (xor masks stay within each 16-lane half).
    float p = hNew * wdV;
    p += __shfl_xor(p, 1, 32);
    p += __shfl_xor(p, 2, 32);
    p += __shfl_xor(p, 4, 32);
    p += __shfl_xor(p, 8, 32);
    if (ml == 0) atomicAdd(&pred[(size_t)b * S_SEQ + t], p);
  }
}

// Copy final H, C into the output tuple region.
__global__ void finalize_kernel(const float* __restrict__ Hf, const float* __restrict__ Cf,
                                float* __restrict__ out) {
  int i = blockIdx.x * blockDim.x + threadIdx.x;
  if (i < 64 * H_DIM) {
    out[64 * S_SEQ + i] = Hf[i];                 // Hf after pred (64*512)
    out[64 * S_SEQ + 64 * H_DIM + i] = Cf[i];    // Cf after Hf
  }
}

extern "C" void kernel_launch(void* const* d_in, const int* in_sizes, int n_in,
                              void* d_out, int out_size, void* d_ws, size_t ws_size,
                              hipStream_t stream) {
  const float* X   = (const float*)d_in[0];
  const float* H0  = (const float*)d_in[1];
  const float* C0  = (const float*)d_in[2];
  const float* Wxi = (const float*)d_in[3];
  const float* Whi = (const float*)d_in[4];
  const float* bI  = (const float*)d_in[5];
  const float* Wxf = (const float*)d_in[6];
  const float* Whf = (const float*)d_in[7];
  const float* bF  = (const float*)d_in[8];
  const float* Wxo = (const float*)d_in[9];
  const float* Who = (const float*)d_in[10];
  const float* bO  = (const float*)d_in[11];
  const float* Wxc = (const float*)d_in[12];
  const float* Whc = (const float*)d_in[13];
  const float* bC  = (const float*)d_in[14];
  const float* Wd  = (const float*)d_in[15];
  const float* bd  = (const float*)d_in[16];
  float* out = (float*)d_out;

  // Workspace layout (~10.8 MB total)
  unsigned short* wxF = (unsigned short*)d_ws;          // 4*GSX bf16 = 2 MB
  unsigned short* whF = wxF + (size_t)4 * GSX;          // 4*GSH bf16 = 8 MB
  float* Hb0 = (float*)(whF + (size_t)4 * GSH);         // 256 KB
  float* Hb1 = Hb0 + 64 * H_DIM;                        // 256 KB
  float* Cb  = Hb1 + 64 * H_DIM;                        // 256 KB

  // 1) Repack weights to bf16 WMMA B-fragments (one-time per call; tiny cost)
  {
    long totX = (long)4 * GSX;
    long totH = (long)4 * GSH;
    pack_b_kernel<<<(unsigned)((totX + 255) / 256), 256, 0, stream>>>(Wxi, Wxf, Wxo, Wxc, wxF, KCX, totX);
    pack_b_kernel<<<(unsigned)((totH + 255) / 256), 256, 0, stream>>>(Whi, Whf, Who, Whc, whF, KCH, totH);
  }

  // 2) Initialize state + pred bias
  init_state_kernel<<<256, 256, 0, stream>>>(H0, C0, bd, Hb0, Cb, out);

  // 3) 512 recurrent steps; stream ordering provides the inter-step barrier.
  for (int t = 0; t < S_SEQ; ++t) {
    const float* Hin = (t & 1) ? Hb1 : Hb0;
    float*       Hou = (t & 1) ? Hb0 : Hb1;
    lstm_step_kernel<<<dim3(16, 4), 128, 0, stream>>>(
        X, wxF, whF, bI, bF, bO, bC, Wd, Hin, Hou, Cb, out, t);
  }

  // 4) Final H lands in Hb0 after an even number of steps.
  finalize_kernel<<<256, 256, 0, stream>>>(Hb0, Cb, out);
}
